// MultiModalAttenNetworkLayers_58755152609593
// MI455X (gfx1250) — compile-verified, matrix operand order checked by reading, not run on hardware
//
#include <hip/hip_runtime.h>
#include <math.h>

#define NNODE 10000
#define NEDGE 100000

typedef __attribute__((ext_vector_type(16))) __bf16 v16bf;
typedef __attribute__((ext_vector_type(8)))  float  v8f;

struct alignas(16) US8 { unsigned short s[8]; };

__device__ __forceinline__ unsigned short f2bf(float f) {
  unsigned int x = __float_as_uint(f);
  unsigned int r = x + 0x7FFFu + ((x >> 16) & 1u);
  return (unsigned short)(r >> 16);
}
__device__ __forceinline__ float bf2f(unsigned short u) {
  return __uint_as_float(((unsigned int)u) << 16);
}
// order-preserving float<->uint encoding for atomicMax-based segment_max
__device__ __forceinline__ unsigned int fenc(float f) {
  unsigned int u = __float_as_uint(f);
  return (u & 0x80000000u) ? ~u : (u | 0x80000000u);
}
__device__ __forceinline__ float fdec(unsigned int u) {
  unsigned int b = (u & 0x80000000u) ? (u ^ 0x80000000u) : ~u;
  return __uint_as_float(b);
}

// A/B fragment loader per CDNA5 ISA 16-bit 16x32 layout:
// lanes 0-15: m=lane, K = 0..7 , 16..23 ; lanes 16-31: m=lane-16, K = 8..15, 24..31
// -> two contiguous 16B chunks per lane. B uses identical pattern on B^T (NxK).
__device__ __forceinline__ v16bf load_frag16(const unsigned short* base, int ldm) {
  int lane = threadIdx.x & 31;
  int m = lane & 15, hi = lane >> 4;
  union { v16bf v; US8 h[2]; } u;
  const unsigned short* row = base + (size_t)m * ldm + hi * 8;
  u.h[0] = *(const US8*)(row);
  u.h[1] = *(const US8*)(row + 16);
  return u.v;
}

__device__ __forceinline__ v8f wmma_bf16(v16bf a, v16bf b, v8f c) {
  return __builtin_amdgcn_wmma_f32_16x16x32_bf16(false, a, false, b, (short)0, c, false, false);
}

// ---------------------------------------------------------------------------
// Generic tiled GEMM: C(MxNout) = act(A(MxK) * B(KxNout) + bias), B given as
// B^T (Nout x K) bf16.
// Block: 256 thr = 8 waves; tile 256(M) x 64(N), BK=32; each wave owns 32 M-rows
// (2 A-frags reused over 4 B-frags -> 8 WMMAs/K-step). Double-buffered LDS,
// one barrier per K-step so global staging overlaps WMMA issue.
// ---------------------------------------------------------------------------
template<bool A_F32, bool RELU, bool OUT_BF16>
__global__ __launch_bounds__(256) void gemm_bf16_kernel(
    const void* __restrict__ Ap, int lda, int M, int K,
    const unsigned short* __restrict__ BT, const float* __restrict__ bias,
    void* __restrict__ Cp, int ldc, int Nout)
{
  constexpr int LS = 48; // LDS row stride (ushorts), multiple of 8 -> 16B rows
  __shared__ alignas(16) unsigned short As[2][256 * LS];
  __shared__ alignas(16) unsigned short Bs[2][64 * LS];
  const int tid  = threadIdx.x;
  const int wave = tid >> 5;
  const int lane = tid & 31;
  const int m0 = blockIdx.x * 256;
  const int n0 = blockIdx.y * 64;

  v8f acc[8];
#pragma unroll
  for (int nt = 0; nt < 8; ++nt)
#pragma unroll
    for (int i = 0; i < 8; ++i) acc[nt][i] = 0.0f;

  const int lr = tid >> 3;         // 0..31 (row within pass)
  const int lc = (tid & 7) << 2;   // 0..28 step 4

  // tile stager: A 256x32, B 64x32 into buffer `buf` for k-offset `k0`
  auto stage = [&](int buf, int k0) {
#pragma unroll
    for (int rb = 0; rb < 256; rb += 32) {
      int row  = m0 + rb + lr;
      int srow = row < M ? row : M - 1;
      unsigned short* d = &As[buf][(rb + lr) * LS + lc];
      if constexpr (A_F32) {
        const float* A = (const float*)Ap;
        const float* s = A + (size_t)srow * lda + k0 + lc;
        const float4 v4 = *(const float4*)s;
        d[0] = f2bf(v4.x); d[1] = f2bf(v4.y); d[2] = f2bf(v4.z); d[3] = f2bf(v4.w);
        if (k0 + 32 < K) __builtin_prefetch(s + 32, 0, 1);
      } else {
        const unsigned short* A = (const unsigned short*)Ap;
        const unsigned short* s = A + (size_t)srow * lda + k0 + lc;
        d[0] = s[0]; d[1] = s[1]; d[2] = s[2]; d[3] = s[3];
        if (k0 + 32 < K) __builtin_prefetch(s + 32, 0, 1);
      }
    }
#pragma unroll
    for (int rb = 0; rb < 64; rb += 32) {
      int n  = n0 + rb + lr;
      int sn = n < Nout ? n : Nout - 1;
      const unsigned short* s = BT + (size_t)sn * K + k0 + lc;
      unsigned short* d = &Bs[buf][(rb + lr) * LS + lc];
      d[0] = s[0]; d[1] = s[1]; d[2] = s[2]; d[3] = s[3];
    }
  };

  stage(0, 0);
  __syncthreads();

  for (int k0 = 0; k0 < K; k0 += 32) {
    const int cur = (k0 >> 5) & 1;
    if (k0 + 32 < K) stage(cur ^ 1, k0 + 32);  // overlap staging with WMMA
    v16bf a0 = load_frag16(&As[cur][(wave * 32) * LS], LS);
    v16bf a1 = load_frag16(&As[cur][(wave * 32 + 16) * LS], LS);
#pragma unroll
    for (int nt = 0; nt < 4; ++nt) {
      v16bf b = load_frag16(&Bs[cur][nt * 16 * LS], LS);
      acc[nt]     = wmma_bf16(a0, b, acc[nt]);
      acc[4 + nt] = wmma_bf16(a1, b, acc[4 + nt]);
    }
    __syncthreads();  // reads(cur) done before cur is restaged next iter
  }

  // epilogue: C/D layout lane -> n = lane&15, m = (lane>>4)*8 + r
  const int hi = lane >> 4;
  const int nl = lane & 15;
#pragma unroll
  for (int ms = 0; ms < 2; ++ms) {
#pragma unroll
    for (int nt = 0; nt < 4; ++nt) {
      int col = n0 + nt * 16 + nl;
      if (col >= Nout) continue;
      float bb = bias[col];
#pragma unroll
      for (int rr = 0; rr < 8; ++rr) {
        int row = m0 + wave * 32 + ms * 16 + hi * 8 + rr;
        if (row >= M) continue;
        float v = acc[ms * 4 + nt][rr] + bb;
        if (RELU) v = fmaxf(v, 0.0f);
        if constexpr (OUT_BF16)
          ((unsigned short*)Cp)[(size_t)row * ldc + col] = f2bf(v);
        else
          ((float*)Cp)[(size_t)row * ldc + col] = v;
      }
    }
  }
}

// ---------------------------------------------------------------------------
// Prep kernels
// ---------------------------------------------------------------------------
__global__ void convT_kernel(const float* __restrict__ w, unsigned short* __restrict__ wt,
                             int K, int Nn) {
  int idx = blockIdx.x * 256 + threadIdx.x;
  if (idx < K * Nn) {
    int n = idx / K, k = idx - n * K;
    wt[idx] = f2bf(w[(size_t)k * Nn + n]);
  }
}

// stage-A fused B^T: [wq(256) | wv(256) | We1[0:256,:](512) | We1[512:768,:](512)], K=256
__global__ void conv_stageA_kernel(const float* __restrict__ wq, const float* __restrict__ wv,
                                   const float* __restrict__ we1, unsigned short* __restrict__ wt) {
  int idx = blockIdx.x * 256 + threadIdx.x;
  if (idx >= 1536 * 256) return;
  int n = idx >> 8, k = idx & 255;
  float v;
  if (n < 256)       v = wq[(size_t)k * 256 + n];
  else if (n < 512)  v = wv[(size_t)k * 256 + (n - 256)];
  else if (n < 1024) v = we1[(size_t)k * 512 + (n - 512)];
  else               v = we1[(size_t)(512 + k) * 512 + (n - 1024)];
  wt[idx] = f2bf(v);
}

// stage-B fused B^T: [wk(256) | We1[256:512,:](512)], K=256
__global__ void conv_stageB_kernel(const float* __restrict__ wk_, const float* __restrict__ we1,
                                   unsigned short* __restrict__ wt) {
  int idx = blockIdx.x * 256 + threadIdx.x;
  if (idx >= 768 * 256) return;
  int n = idx >> 8, k = idx & 255;
  float v = (n < 256) ? wk_[(size_t)k * 256 + n]
                      : we1[(size_t)(256 + k) * 512 + (n - 256)];
  wt[idx] = f2bf(v);
}

__global__ void init_kernel(float* __restrict__ biasA, float* __restrict__ biasB,
                            unsigned int* __restrict__ aggu, float* __restrict__ kl,
                            const float* __restrict__ bq, const float* __restrict__ bv,
                            const float* __restrict__ bk) {
  int idx = blockIdx.x * 256 + threadIdx.x;
  if (idx < NNODE * 256) aggu[idx] = 0u;
  if (idx < 1536) biasA[idx] = (idx < 256) ? bq[idx] : ((idx < 512) ? bv[idx - 256] : 0.0f);
  if (idx < 768)  biasB[idx] = (idx < 256) ? bk[idx] : 0.0f;
  if (idx == 0)   *kl = 0.0f;
}

// h1 = relu(xa[src] + eh + xc[dst] + b_e1), written in-place into stageB cols 256..767
__global__ void h1_kernel(unsigned short* __restrict__ stageB,
                          const unsigned short* __restrict__ stageA,
                          const int* __restrict__ ei, const float* __restrict__ b_e1) {
  size_t idx = (size_t)blockIdx.x * 256 + threadIdx.x;
  if (idx >= (size_t)NEDGE * 512) return;
  int e = (int)(idx >> 9);
  int j = (int)(idx & 511);
  int s = ei[e], d = ei[NEDGE + e];
  size_t o = (size_t)e * 768 + 256 + j;
  float v = bf2f(stageA[(size_t)s * 1536 + 512 + j]) + bf2f(stageB[o])
          + bf2f(stageA[(size_t)d * 1536 + 1024 + j]) + b_e1[j];
  stageB[o] = f2bf(fmaxf(v, 0.0f));
}

// catu = bf16([x | decode(agg)]) ; empty segments -> 0
__global__ void cat_kernel(const float* __restrict__ x, const unsigned int* __restrict__ aggu,
                           unsigned short* __restrict__ catu) {
  size_t idx = (size_t)blockIdx.x * 256 + threadIdx.x;
  if (idx >= (size_t)NNODE * 512) return;
  int n = (int)(idx >> 9), j = (int)(idx & 511);
  float v;
  if (j < 256) v = x[(size_t)n * 256 + j];
  else { unsigned int u = aggu[(size_t)n * 256 + (j - 256)]; v = u ? fdec(u) : 0.0f; }
  catu[idx] = f2bf(v);
}

// ---------------------------------------------------------------------------
// Fused attention: per (edge,head) row: MLPs via WMMA, softmax, KL, value max-scatter
// rows r = e*8+h, block = 4 waves = 64 rows (E*H = 800000 divisible by 64)
// ---------------------------------------------------------------------------
__global__ __launch_bounds__(128) void atten_kernel(
    const unsigned short* __restrict__ stageA,  // N x 1536 [xq|xv|xa|xc]
    const unsigned short* __restrict__ stageB,  // E x 768  [ek|h1]
    const unsigned short* __restrict__ tproj,   // E x 256
    const int* __restrict__ ei,
    const unsigned short* __restrict__ w1t3, const unsigned short* __restrict__ w2t3, // 64x64, 32x64
    const unsigned short* __restrict__ w1tt, const unsigned short* __restrict__ w2tt, // 96x96, 32x96
    const float* __restrict__ b13, const float* __restrict__ b23,
    const float* __restrict__ b1t, const float* __restrict__ b2t,
    float* __restrict__ p3out, float* __restrict__ ptout,
    float* __restrict__ klout, unsigned int* __restrict__ aggu)
{
  __shared__ alignas(16) unsigned short Ain[4][16 * 96];
  __shared__ alignas(16) unsigned short Hid[4][16 * 96];
  __shared__ alignas(16) float Att[4][16 * 64];   // [.. 0..31]=att3, [32..63]=attt
  const int wave = threadIdx.x >> 5;
  const int lane = threadIdx.x & 31;
  const int hi = lane >> 4;
  const int nl = lane & 15;
  const int row0 = blockIdx.x * 64 + wave * 16;

  // gather [q|k|t] rows (stride-8 head interleave) into LDS
  for (int idx = lane; idx < 16 * 96; idx += 32) {
    int rr = idx / 96, cc = idx - rr * 96;
    int grow = row0 + rr;
    int e = grow >> 3, h = grow & 7;
    int s = ei[e];
    unsigned short val;
    if (cc < 32)      val = stageA[(size_t)s * 1536 + (cc << 3) + h];
    else if (cc < 64) val = stageB[(size_t)e * 768 + ((cc - 32) << 3) + h];
    else              val = tproj[(size_t)e * 256 + ((cc - 64) << 3) + h];
    Ain[wave][idx] = val;
  }
  __syncthreads();

  { // text MLP layer 1: Hid = relu(Ain(16x96) @ w1tt^T + b1t)
    v8f acc[6];
#pragma unroll
    for (int nt = 0; nt < 6; ++nt)
#pragma unroll
      for (int i = 0; i < 8; ++i) acc[nt][i] = 0.0f;
#pragma unroll
    for (int kk = 0; kk < 3; ++kk) {
      v16bf a = load_frag16(&Ain[wave][kk * 32], 96);
#pragma unroll
      for (int nt = 0; nt < 6; ++nt) {
        v16bf b = load_frag16(w1tt + nt * 16 * 96 + kk * 32, 96);
        acc[nt] = wmma_bf16(a, b, acc[nt]);
      }
    }
#pragma unroll
    for (int nt = 0; nt < 6; ++nt) {
      int col = nt * 16 + nl;
      float bb = b1t[col];
#pragma unroll
      for (int rr = 0; rr < 8; ++rr)
        Hid[wave][(hi * 8 + rr) * 96 + col] = f2bf(fmaxf(acc[nt][rr] + bb, 0.0f));
    }
  }
  __syncthreads();
  { // text MLP layer 2 -> Att cols 32..63
    v8f acc[2];
#pragma unroll
    for (int nt = 0; nt < 2; ++nt)
#pragma unroll
      for (int i = 0; i < 8; ++i) acc[nt][i] = 0.0f;
#pragma unroll
    for (int kk = 0; kk < 3; ++kk) {
      v16bf a = load_frag16(&Hid[wave][kk * 32], 96);
#pragma unroll
      for (int nt = 0; nt < 2; ++nt) {
        v16bf b = load_frag16(w2tt + nt * 16 * 96 + kk * 32, 96);
        acc[nt] = wmma_bf16(a, b, acc[nt]);
      }
    }
#pragma unroll
    for (int nt = 0; nt < 2; ++nt) {
      int col = nt * 16 + nl;
      float bb = b2t[col];
#pragma unroll
      for (int rr = 0; rr < 8; ++rr)
        Att[wave][(hi * 8 + rr) * 64 + 32 + col] = acc[nt][rr] + bb;
    }
  }
  __syncthreads();
  { // 3d MLP layer 1: uses Ain cols 0..63
    v8f acc[4];
#pragma unroll
    for (int nt = 0; nt < 4; ++nt)
#pragma unroll
      for (int i = 0; i < 8; ++i) acc[nt][i] = 0.0f;
#pragma unroll
    for (int kk = 0; kk < 2; ++kk) {
      v16bf a = load_frag16(&Ain[wave][kk * 32], 96);
#pragma unroll
      for (int nt = 0; nt < 4; ++nt) {
        v16bf b = load_frag16(w1t3 + nt * 16 * 64 + kk * 32, 64);
        acc[nt] = wmma_bf16(a, b, acc[nt]);
      }
    }
#pragma unroll
    for (int nt = 0; nt < 4; ++nt) {
      int col = nt * 16 + nl;
      float bb = b13[col];
#pragma unroll
      for (int rr = 0; rr < 8; ++rr)
        Hid[wave][(hi * 8 + rr) * 96 + col] = f2bf(fmaxf(acc[nt][rr] + bb, 0.0f));
    }
  }
  __syncthreads();
  { // 3d MLP layer 2 -> Att cols 0..31
    v8f acc[2];
#pragma unroll
    for (int nt = 0; nt < 2; ++nt)
#pragma unroll
      for (int i = 0; i < 8; ++i) acc[nt][i] = 0.0f;
#pragma unroll
    for (int kk = 0; kk < 2; ++kk) {
      v16bf a = load_frag16(&Hid[wave][kk * 32], 96);
#pragma unroll
      for (int nt = 0; nt < 2; ++nt) {
        v16bf b = load_frag16(w2t3 + nt * 16 * 64 + kk * 32, 64);
        acc[nt] = wmma_bf16(a, b, acc[nt]);
      }
    }
#pragma unroll
    for (int nt = 0; nt < 2; ++nt) {
      int col = nt * 16 + nl;
      float bb = b23[col];
#pragma unroll
      for (int rr = 0; rr < 8; ++rr)
        Att[wave][(hi * 8 + rr) * 64 + col] = acc[nt][rr] + bb;
    }
  }
  __syncthreads();

  // softmax / KL / value scatter: one row per thread (threads 0..63)
  const int t = threadIdx.x;
  if (t < 64) {
    const int w = t >> 4, mr = t & 15;
    const int grow = blockIdx.x * 64 + t;
    const int e = grow >> 3, h = grow & 7;
    const float* arow = &Att[w][mr * 64];
    const float inv_temp = 0.17677669529663687f;  // 1/sqrt(32)
    float p3[32], pt[32];
    float m1 = -3.0e38f, m2 = -3.0e38f;
#pragma unroll
    for (int f = 0; f < 32; ++f) { m1 = fmaxf(m1, arow[f]); m2 = fmaxf(m2, arow[32 + f]); }
    float s1 = 0.f, s2 = 0.f;
#pragma unroll
    for (int f = 0; f < 32; ++f) {
      p3[f] = __expf((arow[f] - m1) * inv_temp);       s1 += p3[f];
      pt[f] = __expf((arow[32 + f] - m2) * inv_temp);  s2 += pt[f];
    }
    float r1 = 1.0f / s1, r2 = 1.0f / s2;
#pragma unroll
    for (int f = 0; f < 32; ++f) { p3[f] *= r1; pt[f] *= r2; }
    // lsm = log_softmax(p3); lt = log_softmax(pt); tgt = exp(lt)
    float m3 = -3.0e38f, mt2 = -3.0e38f;
#pragma unroll
    for (int f = 0; f < 32; ++f) { m3 = fmaxf(m3, p3[f]); mt2 = fmaxf(mt2, pt[f]); }
    float s3 = 0.f, st = 0.f;
#pragma unroll
    for (int f = 0; f < 32; ++f) { s3 += __expf(p3[f] - m3); st += __expf(pt[f] - mt2); }
    float c3 = m3 + __logf(s3);
    float ct = mt2 + __logf(st);
    float kl = 0.f;
#pragma unroll
    for (int f = 0; f < 32; ++f) {
      float lt = pt[f] - ct;
      float tg = __expf(lt);
      kl += tg * (lt - (p3[f] - c3));
    }
    atomicAdd(klout, kl * (1.0f / 800000.0f));

    int s = ei[e], d2 = ei[NEDGE + e];
    size_t ob = (size_t)e * 256;
#pragma unroll
    for (int f = 0; f < 32; ++f) {
      int cc = (f << 3) + h;
      p3out[ob + cc] = p3[f];
      ptout[ob + cc] = pt[f];
      float vv = bf2f(stageA[(size_t)d2 * 1536 + 256 + cc]);   // v = xv[dst]
      atomicMax(&aggu[(size_t)s * 256 + cc], fenc(p3[f] * vv));
    }
  }
}

// ---------------------------------------------------------------------------
extern "C" void kernel_launch(void* const* d_in, const int* in_sizes, int n_in,
                              void* d_out, int out_size, void* d_ws, size_t ws_size,
                              hipStream_t stream) {
  (void)in_sizes; (void)n_in; (void)out_size;
  const float* x      = (const float*)d_in[0];
  const float* ef     = (const float*)d_in[1];
  const float* te     = (const float*)d_in[2];
  const int*   ei     = (const int*)d_in[3];
  const float* w_text = (const float*)d_in[4];
  const float* b_text = (const float*)d_in[5];
  const float* w_e1   = (const float*)d_in[6];
  const float* b_e1   = (const float*)d_in[7];
  const float* w_e2   = (const float*)d_in[8];
  const float* b_e2   = (const float*)d_in[9];
  const float* wq     = (const float*)d_in[10];
  const float* bq     = (const float*)d_in[11];
  const float* wk_    = (const float*)d_in[12];
  const float* bk     = (const float*)d_in[13];
  const float* wv     = (const float*)d_in[14];
  const float* bv     = (const float*)d_in[15];
  const float* wa31   = (const float*)d_in[16];
  const float* ba31   = (const float*)d_in[17];
  const float* wa32   = (const float*)d_in[18];
  const float* ba32   = (const float*)d_in[19];
  const float* wat1   = (const float*)d_in[20];
  const float* bat1   = (const float*)d_in[21];
  const float* wat2   = (const float*)d_in[22];
  const float* bat2   = (const float*)d_in[23];
  const float* wu1    = (const float*)d_in[24];
  const float* bu1    = (const float*)d_in[25];
  const float* wu2    = (const float*)d_in[26];
  const float* bu2    = (const float*)d_in[27];

  float* out_node = (float*)d_out;                 // N*256
  float* out_edge = out_node + 2560000;            // E*256
  float* out_p3   = out_node + 28160000;           // E*256
  float* out_pt   = out_node + 53760000;           // E*256
  float* out_kl   = out_node + 79360000;           // 1

  char* ws = (char*)d_ws;
  size_t off = 0;
  auto alloc = [&](size_t bytes) -> void* {
    void* p = ws + off;
    off += (bytes + 255) & ~(size_t)255;
    return p;
  };
  unsigned short* wtA    = (unsigned short*)alloc((size_t)1536 * 256 * 2);
  unsigned short* wtB    = (unsigned short*)alloc((size_t)768 * 256 * 2);
  unsigned short* wtText = (unsigned short*)alloc((size_t)256 * 512 * 2);
  unsigned short* wtE2   = (unsigned short*)alloc((size_t)256 * 512 * 2);
  unsigned short* wtA31  = (unsigned short*)alloc((size_t)64 * 64 * 2);
  unsigned short* wtA32  = (unsigned short*)alloc((size_t)32 * 64 * 2);
  unsigned short* wtT1   = (unsigned short*)alloc((size_t)96 * 96 * 2);
  unsigned short* wtT2   = (unsigned short*)alloc((size_t)32 * 96 * 2);
  unsigned short* wtU1   = (unsigned short*)alloc((size_t)512 * 512 * 2);
  unsigned short* wtU2   = (unsigned short*)alloc((size_t)256 * 512 * 2);
  float*          biasA  = (float*)alloc(1536 * 4);
  float*          biasB  = (float*)alloc(768 * 4);
  unsigned short* stA    = (unsigned short*)alloc((size_t)NNODE * 1536 * 2);
  unsigned short* stB    = (unsigned short*)alloc((size_t)NEDGE * 768 * 2);
  unsigned short* tproj  = (unsigned short*)alloc((size_t)NEDGE * 256 * 2);
  unsigned int*   aggu   = (unsigned int*)alloc((size_t)NNODE * 256 * 4);
  unsigned short* catu   = (unsigned short*)alloc((size_t)NNODE * 512 * 2);
  unsigned short* hu     = (unsigned short*)alloc((size_t)NNODE * 512 * 2);
  if (off > ws_size) return;  // insufficient workspace

  auto cd = [](long a, long b) -> unsigned { return (unsigned)((a + b - 1) / b); };

  // init + weight prep
  init_kernel<<<10000, 256, 0, stream>>>(biasA, biasB, aggu, out_kl, bq, bv, bk);
  conv_stageA_kernel<<<1536, 256, 0, stream>>>(wq, wv, w_e1, wtA);
  conv_stageB_kernel<<<768, 256, 0, stream>>>(wk_, w_e1, wtB);
  convT_kernel<<<cd(512 * 256, 256), 256, 0, stream>>>(w_text, wtText, 512, 256);
  convT_kernel<<<cd(512 * 256, 256), 256, 0, stream>>>(w_e2, wtE2, 512, 256);
  convT_kernel<<<cd(64 * 64, 256), 256, 0, stream>>>(wa31, wtA31, 64, 64);
  convT_kernel<<<cd(64 * 32, 256), 256, 0, stream>>>(wa32, wtA32, 64, 32);
  convT_kernel<<<cd(96 * 96, 256), 256, 0, stream>>>(wat1, wtT1, 96, 96);
  convT_kernel<<<cd(96 * 32, 256), 256, 0, stream>>>(wat2, wtT2, 96, 32);
  convT_kernel<<<cd(512 * 512, 256), 256, 0, stream>>>(wu1, wtU1, 512, 512);
  convT_kernel<<<cd(512 * 256, 256), 256, 0, stream>>>(wu2, wtU2, 512, 256);

  // stage A: per-node projections [xq|xv|xa|xc] = x @ [wq|wv|We1_lo|We1_hi]
  dim3 gA(cd(NNODE, 256), 1536 / 64);
  gemm_bf16_kernel<true, false, true><<<gA, 256, 0, stream>>>(
      x, 256, NNODE, 256, wtA, biasA, stA, 1536, 1536);
  // stage B: per-edge [ek|eh] = edge @ [wk|We1_mid]
  dim3 gB(cd(NEDGE, 256), 768 / 64);
  gemm_bf16_kernel<true, false, true><<<gB, 256, 0, stream>>>(
      ef, 256, NEDGE, 256, wtB, biasB, stB, 768, 768);
  // t_proj = text @ w_text + b_text
  dim3 gT(cd(NEDGE, 256), 256 / 64);
  gemm_bf16_kernel<true, false, true><<<gT, 256, 0, stream>>>(
      te, 512, NEDGE, 512, wtText, b_text, tproj, 256, 256);
  // h1 = relu(xa[src] + eh + xc[dst] + b_e1) (in place in stB cols 256..767)
  h1_kernel<<<cd((long)NEDGE * 512, 256), 256, 0, stream>>>(stB, stA, ei, b_e1);
  // edge_out = relu(h1 @ w_e2 + b_e2)
  gemm_bf16_kernel<false, true, false><<<gT, 256, 0, stream>>>(
      stB + 256, 768, NEDGE, 512, wtE2, b_e2, out_edge, 256, 256);
  // attention MLPs + softmax + KL + value max-scatter
  atten_kernel<<<12500, 128, 0, stream>>>(stA, stB, tproj, ei,
      wtA31, wtA32, wtT1, wtT2, ba31, ba32, bat1, bat2,
      out_p3, out_pt, out_kl, aggu);
  // node update
  cat_kernel<<<cd((long)NNODE * 512, 256), 256, 0, stream>>>(x, aggu, catu);
  dim3 gU1(cd(NNODE, 256), 512 / 64);
  gemm_bf16_kernel<false, true, true><<<gU1, 256, 0, stream>>>(
      catu, 512, NNODE, 512, wtU1, bu1, hu, 512, 512);
  dim3 gU2(cd(NNODE, 256), 256 / 64);
  gemm_bf16_kernel<false, true, false><<<gU2, 256, 0, stream>>>(
      hu, 512, NNODE, 512, wtU2, bu2, out_node, 256, 256);
}
